// LinearAttention_5488968204421
// MI455X (gfx1250) — compile-verified, matrix operand order checked by reading, not run on hardware
//
#include <hip/hip_runtime.h>

// ---------------------------------------------------------------------------
// LinearAttention (performer / l1-softmax) for MI455X (gfx1250, wave32, WMMA)
// B=4, T=8192, C=1024, H=8, D=128
// bf16 v_wmma_f32_16x16x32_bf16 for all matrix math, fp32 for softmax /
// reductions / accumulation / residual. Async global->LDS (ASYNCcnt) for the
// straight-copy tiles when the toolchain exposes the gfx1250 builtins.
// ---------------------------------------------------------------------------

constexpr int BB = 4;
constexpr int TT = 8192;
constexpr int CC = 1024;
constexpr int HH = 8;
constexpr int DD = 128;

typedef __attribute__((ext_vector_type(16))) __bf16 v16bf;
typedef __attribute__((ext_vector_type(8)))  float  v8f;
typedef __attribute__((ext_vector_type(4)))  int    v4i;

// --- gfx1250 async global->LDS copy (ASYNCcnt), guarded ---------------------
#if defined(__has_builtin)
#if __has_builtin(__builtin_amdgcn_global_load_async_to_lds_b128) && \
    __has_builtin(__builtin_amdgcn_s_wait_asynccnt)
#define USE_ASYNC_LDS 1
#endif
#endif

__device__ __forceinline__ void copy16_g2l(const unsigned short* g,
                                           unsigned short* l) {
#if defined(USE_ASYNC_LDS)
  __builtin_amdgcn_global_load_async_to_lds_b128(
      (__attribute__((address_space(1))) v4i*)g,
      (__attribute__((address_space(3))) v4i*)l, 0, 0);
#else
  *(uint4*)l = *(const uint4*)g;
#endif
}
__device__ __forceinline__ void wait_async_copies() {
#if defined(USE_ASYNC_LDS)
  __builtin_amdgcn_s_wait_asynccnt(0);
#endif
}

// --- bf16 <-> f32 helpers (bit-level, round-to-nearest-even) ---------------
__device__ __forceinline__ unsigned short f2bs(float f) {
  unsigned u = __builtin_bit_cast(unsigned, f);
  u += 0x7FFFu + ((u >> 16) & 1u);
  return (unsigned short)(u >> 16);
}
__device__ __forceinline__ float bs2f(unsigned short s) {
  unsigned u = ((unsigned)s) << 16;
  return __builtin_bit_cast(float, u);
}

// --- WMMA fragment load from an LDS tile stored [row][k], k contiguous -----
// 16-bit A/B layout (ISA 7.12.2): lane<16 -> row=lane,   K in {0..7,16..23}
//                                 lane>=16 -> row=lane-16, K in {8..15,24..31}
struct F2 { uint4 a, b; };
__device__ __forceinline__ v16bf load_frag(const unsigned short* base, int ld,
                                           int row0, int lane) {
  const unsigned short* p =
      base + (size_t)(row0 + (lane & 15)) * ld + ((lane >> 4) * 8);
  F2 f;
  f.a = *(const uint4*)(p);       // K 0..7   (or 8..15 for upper lanes)
  f.b = *(const uint4*)(p + 16);  // K 16..23 (or 24..31)
  return __builtin_bit_cast(v16bf, f);
}

// ---------------------------------------------------------------------------
// Generic GEMM: C[M,N] = A[M,K](bf16) * W[K,N](bf16) + bias(f32)
// Block tile 128x128, BK=32, 256 threads = 8 waves in a 2x4 grid,
// each wave owns a 64x32 output patch = 4x2 WMMA tiles.
// ---------------------------------------------------------------------------
#define LDT 40  // LDS row stride in ushort (80 B = 5 x 16 B, aligned + staggered)

template <bool OUT_BF16>
__global__ __launch_bounds__(256) void gemm_bf16_kernel(
    const unsigned short* __restrict__ A, const unsigned short* __restrict__ W,
    const float* __restrict__ bias, float* __restrict__ Cf,
    unsigned short* __restrict__ Cb, int M, int N, int K) {
  __shared__ unsigned short As[128 * LDT];  // [m][k]
  __shared__ unsigned short Bs[128 * LDT];  // [n][k] (transposed on store)

  const int tid = threadIdx.x;
  const int lane = tid & 31;
  const int wid = tid >> 5;
  const int wm = (wid >> 2) * 64;
  const int wn = (wid & 3) * 32;
  const int gm0 = blockIdx.x * 128;
  const int gn0 = blockIdx.y * 128;

  v8f acc[4][2] = {};

  for (int k0 = 0; k0 < K; k0 += 32) {
    // A tile: 128 rows x 32 halfs = 512 x 16 B, async copy global->LDS
    for (int i = tid; i < 512; i += 256) {
      int r = i >> 2, kc = (i & 3) * 8;
      copy16_g2l(&A[(size_t)(gm0 + r) * K + k0 + kc], &As[r * LDT + kc]);
    }
    // W tile: 32 k-rows x 128 n, read coalesced, scatter transposed to Bs[n][k]
    for (int i = tid; i < 512; i += 256) {
      int kr = i >> 4, nc = (i & 15) * 8;
      uint4 w = *(const uint4*)&W[(size_t)(k0 + kr) * N + gn0 + nc];
      const unsigned short* s = (const unsigned short*)&w;
#pragma unroll
      for (int j = 0; j < 8; j++) Bs[(nc + j) * LDT + kr] = s[j];
    }
    wait_async_copies();
    __syncthreads();

    v16bf af[4], bf[2];
#pragma unroll
    for (int mi = 0; mi < 4; mi++) af[mi] = load_frag(As, LDT, wm + mi * 16, lane);
#pragma unroll
    for (int ni = 0; ni < 2; ni++) bf[ni] = load_frag(Bs, LDT, wn + ni * 16, lane);
#pragma unroll
    for (int mi = 0; mi < 4; mi++)
#pragma unroll
      for (int ni = 0; ni < 2; ni++)
        acc[mi][ni] = __builtin_amdgcn_wmma_f32_16x16x32_bf16(
            false, af[mi], false, bf[ni], (short)0, acc[mi][ni], false, false);
    __syncthreads();
  }

  // Epilogue: C/D layout: VGPR i -> M = i (+8 for lanes 16..31), N = lane&15
  const int rhalf = (lane >> 4) * 8;
  const int col = lane & 15;
#pragma unroll
  for (int mi = 0; mi < 4; mi++)
#pragma unroll
    for (int ni = 0; ni < 2; ni++) {
      int gc = gn0 + wn + ni * 16 + col;
      float bv = bias ? bias[gc] : 0.0f;
#pragma unroll
      for (int i = 0; i < 8; i++) {
        int gr = gm0 + wm + mi * 16 + i + rhalf;
        float val = acc[mi][ni][i] + bv;
        if constexpr (OUT_BF16)
          Cb[(size_t)gr * N + gc] = f2bs(val);
        else
          Cf[(size_t)gr * N + gc] = val;
      }
    }
}

// ---------------------------------------------------------------------------
// In-place softmax over contiguous 128-element head vectors (one wave each).
// ---------------------------------------------------------------------------
__global__ __launch_bounds__(256) void softmax128_kernel(
    unsigned short* __restrict__ t, int nChunks) {
  int lane = threadIdx.x & 31, wid = threadIdx.x >> 5;
  int chunk = blockIdx.x * 8 + wid;
  if (chunk >= nChunks) return;
  unsigned short* p = t + (size_t)chunk * 128 + lane * 4;
  uint2 raw = *(uint2*)p;
  unsigned short s[4];
  __builtin_memcpy(s, &raw, 8);
  float f[4];
#pragma unroll
  for (int j = 0; j < 4; j++) f[j] = bs2f(s[j]);
  float m = fmaxf(fmaxf(f[0], f[1]), fmaxf(f[2], f[3]));
#pragma unroll
  for (int off = 16; off > 0; off >>= 1) m = fmaxf(m, __shfl_xor(m, off, 32));
  float sum = 0.f;
#pragma unroll
  for (int j = 0; j < 4; j++) { f[j] = __expf(f[j] - m); sum += f[j]; }
#pragma unroll
  for (int off = 16; off > 0; off >>= 1) sum += __shfl_xor(sum, off, 32);
  float inv = 1.0f / sum;
#pragma unroll
  for (int j = 0; j < 4; j++) s[j] = f2bs(f[j] * inv);
  __builtin_memcpy(&raw, s, 8);
  *(uint2*)p = raw;
}

// ---------------------------------------------------------------------------
// k_sum[b,h,d] = sum_t k_softmax[b,t,h,d]   (one block per (b,h), 128 thr)
// ---------------------------------------------------------------------------
__global__ __launch_bounds__(128) void ksum_kernel(
    const unsigned short* __restrict__ kb, float* __restrict__ ksum) {
  int bh = blockIdx.x;
  int b = bh >> 3, h = bh & 7;
  int d = threadIdx.x;
  const unsigned short* p = kb + (size_t)b * TT * CC + h * DD + d;
  float s = 0.f;
  for (int t = 0; t < TT; t++) {
    if ((t & 63) == 0)
      __builtin_prefetch(p + (size_t)(t + 128) * CC, 0, 0);  // global_prefetch_b8
    s += bs2f(p[(size_t)t * CC]);
  }
  ksum[bh * DD + d] = s;
}

// ---------------------------------------------------------------------------
// dinv[bt*H+h] = 1 / dot(q_softmax[bt,h,:], ksum[b,h,:])  (one wave per chunk)
// ---------------------------------------------------------------------------
__global__ __launch_bounds__(256) void dinv_kernel(
    const unsigned short* __restrict__ qb, const float* __restrict__ ksum,
    float* __restrict__ dinv, int nChunks) {
  int lane = threadIdx.x & 31, wid = threadIdx.x >> 5;
  int chunk = blockIdx.x * 8 + wid;
  if (chunk >= nChunks) return;
  int h = chunk & 7;
  int bt = chunk >> 3;
  int b = bt >> 13;  // bt / T, T = 8192
  const unsigned short* p = qb + (size_t)chunk * 128 + lane * 4;
  const float* ks = ksum + (b * HH + h) * DD + lane * 4;
  uint2 raw = *(const uint2*)p;
  unsigned short s[4];
  __builtin_memcpy(s, &raw, 8);
  float dot = 0.f;
#pragma unroll
  for (int j = 0; j < 4; j++) dot += bs2f(s[j]) * ks[j];
#pragma unroll
  for (int off = 16; off > 0; off >>= 1) dot += __shfl_xor(dot, off, 32);
  if (lane == 0) dinv[chunk] = 1.0f / dot;
}

// ---------------------------------------------------------------------------
// context[b,h] += k^T v over a T-chunk of 1024 (WMMA, atomic f32 merge).
// grid: (T/1024, B*H)
// ---------------------------------------------------------------------------
__global__ __launch_bounds__(256) void ctx_kernel(
    const unsigned short* __restrict__ kb, const unsigned short* __restrict__ vb,
    float* __restrict__ ctx) {
  __shared__ unsigned short Ks[128 * LDT];  // [d][t]
  __shared__ unsigned short Vs[128 * LDT];  // [e][t]
  const int tid = threadIdx.x, lane = tid & 31, wid = tid >> 5;
  const int wm = (wid >> 2) * 64, wn = (wid & 3) * 32;
  const int bh = blockIdx.y;
  const int b = bh >> 3, h = bh & 7;
  const int t0 = blockIdx.x * 1024;

  v8f acc[4][2] = {};

  for (int ts = 0; ts < 1024; ts += 32) {
    for (int i = tid; i < 512; i += 256) {  // k tile 32t x 128d, scatter -> Ks[d][t]
      int tr = i >> 4, dc = (i & 15) * 8;
      size_t g = (size_t)(b * TT + t0 + ts + tr) * CC + h * DD + dc;
      uint4 w = *(const uint4*)&kb[g];
      const unsigned short* s = (const unsigned short*)&w;
#pragma unroll
      for (int j = 0; j < 8; j++) Ks[(dc + j) * LDT + tr] = s[j];
    }
    for (int i = tid; i < 512; i += 256) {  // v tile -> Vs[e][t]
      int tr = i >> 4, ec = (i & 15) * 8;
      size_t g = (size_t)(b * TT + t0 + ts + tr) * CC + h * DD + ec;
      uint4 w = *(const uint4*)&vb[g];
      const unsigned short* s = (const unsigned short*)&w;
#pragma unroll
      for (int j = 0; j < 8; j++) Vs[(ec + j) * LDT + tr] = s[j];
    }
    __syncthreads();
    v16bf af[4], bf[2];
#pragma unroll
    for (int mi = 0; mi < 4; mi++) af[mi] = load_frag(Ks, LDT, wm + mi * 16, lane);
#pragma unroll
    for (int ni = 0; ni < 2; ni++) bf[ni] = load_frag(Vs, LDT, wn + ni * 16, lane);
#pragma unroll
    for (int mi = 0; mi < 4; mi++)
#pragma unroll
      for (int ni = 0; ni < 2; ni++)
        acc[mi][ni] = __builtin_amdgcn_wmma_f32_16x16x32_bf16(
            false, af[mi], false, bf[ni], (short)0, acc[mi][ni], false, false);
    __syncthreads();
  }

  float* cp = ctx + (size_t)bh * DD * DD;
  const int rhalf = (lane >> 4) * 8;
  const int col = lane & 15;
#pragma unroll
  for (int mi = 0; mi < 4; mi++)
#pragma unroll
    for (int ni = 0; ni < 2; ni++)
#pragma unroll
      for (int i = 0; i < 8; i++)
        atomicAdd(&cp[(size_t)(wm + mi * 16 + i + rhalf) * DD + wn + ni * 16 + col],
                  acc[mi][ni][i]);
}

// ---------------------------------------------------------------------------
// y = (q @ ctx) * dinv + q  -> bf16 into [B*T, C] head slots.
// grid: (T/128, B*H). Full 128x128 q-tile + ctx-tile staged in LDS (~70 KB).
// ---------------------------------------------------------------------------
#define LDQ 136  // 272 B row stride = 17 x 16 B

__global__ __launch_bounds__(256) void y_kernel(
    const unsigned short* __restrict__ qb, const unsigned short* __restrict__ ctxb,
    const float* __restrict__ dinv, unsigned short* __restrict__ yb) {
  __shared__ unsigned short Qs[128 * LDQ];  // [t][d]
  __shared__ unsigned short Cs[128 * LDQ];  // [e][d] (transposed ctx)
  const int tid = threadIdx.x, lane = tid & 31, wid = tid >> 5;
  const int wm = (wid >> 2) * 64, wn = (wid & 3) * 32;
  const int bh = blockIdx.y;
  const int b = bh >> 3, h = bh & 7;
  const int t0 = blockIdx.x * 128;

  for (int i = tid; i < 2048; i += 256) {  // q tile: async copy global->LDS
    int r = i >> 4, dc = (i & 15) * 8;
    copy16_g2l(&qb[(size_t)(b * TT + t0 + r) * CC + h * DD + dc],
               &Qs[r * LDQ + dc]);
  }
  for (int i = tid; i < 2048; i += 256) {  // ctx [d][e] -> Cs[e][d]
    int d = i >> 4, ec = (i & 15) * 8;
    uint4 w = *(const uint4*)&ctxb[(size_t)bh * DD * DD + (size_t)d * DD + ec];
    const unsigned short* s = (const unsigned short*)&w;
#pragma unroll
    for (int j = 0; j < 8; j++) Cs[(ec + j) * LDQ + d] = s[j];
  }
  wait_async_copies();
  __syncthreads();

  v8f acc[4][2] = {};
#pragma unroll
  for (int k0 = 0; k0 < 128; k0 += 32) {
    v16bf af[4], bf[2];
#pragma unroll
    for (int mi = 0; mi < 4; mi++)
      af[mi] = load_frag(Qs + k0, LDQ, wm + mi * 16, lane);
#pragma unroll
    for (int ni = 0; ni < 2; ni++)
      bf[ni] = load_frag(Cs + k0, LDQ, wn + ni * 16, lane);
#pragma unroll
    for (int mi = 0; mi < 4; mi++)
#pragma unroll
      for (int ni = 0; ni < 2; ni++)
        acc[mi][ni] = __builtin_amdgcn_wmma_f32_16x16x32_bf16(
            false, af[mi], false, bf[ni], (short)0, acc[mi][ni], false, false);
  }

  const int rhalf = (lane >> 4) * 8;
  const int col = lane & 15;
#pragma unroll
  for (int mi = 0; mi < 4; mi++)
#pragma unroll
    for (int ni = 0; ni < 2; ni++) {
      int e = wn + ni * 16 + col;
#pragma unroll
      for (int i = 0; i < 8; i++) {
        int tl = wm + mi * 16 + i + rhalf;
        size_t bt = (size_t)b * TT + t0 + tl;
        float di = dinv[bt * HH + h];
        float val = acc[mi][ni][i] * di + bs2f(Qs[tl * LDQ + e]);
        yb[bt * CC + h * DD + e] = f2bs(val);
      }
    }
}

// ---------------------------------------------------------------------------
// Elementwise utilities
// ---------------------------------------------------------------------------
__global__ void f32_to_bf16_kernel(const float* __restrict__ in,
                                   unsigned short* __restrict__ out, size_t n) {
  for (size_t i = (size_t)blockIdx.x * blockDim.x + threadIdx.x; i < n;
       i += (size_t)gridDim.x * blockDim.x)
    out[i] = f2bs(in[i]);
}
__global__ void zero_f32_kernel(float* __restrict__ p, size_t n) {
  for (size_t i = (size_t)blockIdx.x * blockDim.x + threadIdx.x; i < n;
       i += (size_t)gridDim.x * blockDim.x)
    p[i] = 0.0f;
}

// ---------------------------------------------------------------------------
// Launcher
// ---------------------------------------------------------------------------
extern "C" void kernel_launch(void* const* d_in, const int* in_sizes, int n_in,
                              void* d_out, int out_size, void* d_ws, size_t ws_size,
                              hipStream_t stream) {
  const float* x  = (const float*)d_in[0];
  const float* Wq = (const float*)d_in[1];
  const float* bq = (const float*)d_in[2];
  const float* Wk = (const float*)d_in[3];
  const float* bk = (const float*)d_in[4];
  const float* Wv = (const float*)d_in[5];
  const float* bv = (const float*)d_in[6];
  const float* Wp = (const float*)d_in[7];
  const float* bp = (const float*)d_in[8];
  float* out = (float*)d_out;

  const size_t M = (size_t)BB * TT;       // 32768
  const size_t KC = (size_t)CC;           // 1024
  const size_t nX = M * KC;               // 33.5M
  const size_t nW = KC * KC;              // 1M
  const size_t nCtx = (size_t)BB * HH * DD * DD;  // 524288
  const size_t nChunks = M * HH;          // 262144

  char* ws = (char*)d_ws;
  size_t off = 0;
  auto arena = [&](size_t bytes) -> char* {
    char* r = ws + off;
    off += (bytes + 255) & ~(size_t)255;
    return r;
  };
  unsigned short* xb  = (unsigned short*)arena(nX * 2);
  unsigned short* Wqb = (unsigned short*)arena(nW * 2);
  unsigned short* Wkb = (unsigned short*)arena(nW * 2);
  unsigned short* Wvb = (unsigned short*)arena(nW * 2);
  unsigned short* Wpb = (unsigned short*)arena(nW * 2);
  unsigned short* qb  = (unsigned short*)arena(nX * 2);
  unsigned short* kb  = (unsigned short*)arena(nX * 2);
  unsigned short* vb  = (unsigned short*)arena(nX * 2);
  unsigned short* yb  = (unsigned short*)arena(nX * 2);
  float* ksum = (float*)arena((size_t)BB * HH * DD * 4);
  float* dinv = (float*)arena(nChunks * 4);
  float* ctx  = (float*)arena(nCtx * 4);
  unsigned short* ctxb = (unsigned short*)arena(nCtx * 2);
  (void)ws_size; (void)in_sizes; (void)n_in; (void)out_size;

  // 1) fp32 -> bf16 conversions
  f32_to_bf16_kernel<<<4096, 256, 0, stream>>>(x, xb, nX);
  f32_to_bf16_kernel<<<1024, 256, 0, stream>>>(Wq, Wqb, nW);
  f32_to_bf16_kernel<<<1024, 256, 0, stream>>>(Wk, Wkb, nW);
  f32_to_bf16_kernel<<<1024, 256, 0, stream>>>(Wv, Wvb, nW);
  f32_to_bf16_kernel<<<1024, 256, 0, stream>>>(Wp, Wpb, nW);

  // 2) QKV projections (WMMA GEMM, bf16 out, fused bias)
  dim3 ggrid(M / 128, CC / 128);
  gemm_bf16_kernel<true><<<ggrid, 256, 0, stream>>>(xb, Wqb, bq, nullptr, qb,
                                                    (int)M, CC, CC);
  gemm_bf16_kernel<true><<<ggrid, 256, 0, stream>>>(xb, Wkb, bk, nullptr, kb,
                                                    (int)M, CC, CC);
  gemm_bf16_kernel<true><<<ggrid, 256, 0, stream>>>(xb, Wvb, bv, nullptr, vb,
                                                    (int)M, CC, CC);

  // 3) softmax over head dim (in place on q, k)
  softmax128_kernel<<<nChunks / 8, 256, 0, stream>>>(qb, (int)nChunks);
  softmax128_kernel<<<nChunks / 8, 256, 0, stream>>>(kb, (int)nChunks);

  // 4) k_sum and D_inv
  ksum_kernel<<<BB * HH, 128, 0, stream>>>(kb, ksum);
  dinv_kernel<<<nChunks / 8, 256, 0, stream>>>(qb, ksum, dinv, (int)nChunks);

  // 5) context = k^T v (atomic-merged T-split WMMA), then convert to bf16
  zero_f32_kernel<<<512, 256, 0, stream>>>(ctx, nCtx);
  ctx_kernel<<<dim3(TT / 1024, BB * HH), 256, 0, stream>>>(kb, vb, ctx);
  f32_to_bf16_kernel<<<512, 256, 0, stream>>>(ctx, ctxb, nCtx);

  // 6) y = q @ ctx * D_inv + q
  y_kernel<<<dim3(TT / 128, BB * HH), 256, 0, stream>>>(qb, ctxb, dinv, yb);

  // 7) output projection (fp32 out, fused bias)
  gemm_bf16_kernel<false><<<ggrid, 256, 0, stream>>>(yb, Wpb, bp, out, nullptr,
                                                     (int)M, CC, CC);
}
